// AttentionHead_28741921145456
// MI455X (gfx1250) — matrix operand order repaired and checked by reading, not compiled, not run on hardware
//
#include <hip/hip_runtime.h>
#include <cstdint>
#include <cstddef>

typedef __attribute__((ext_vector_type(16))) _Float16 v16h;
typedef __attribute__((ext_vector_type(8)))  float    v8f;

typedef __attribute__((ext_vector_type(4))) unsigned int u32x4;
typedef __attribute__((ext_vector_type(8))) int          i32x8;
typedef __attribute__((ext_vector_type(4))) int          i32x4;

#define NEG_INF (-__builtin_inff())

// Tensor Data Mover availability (probe-confirmed builtin; arity differs per toolchain)
#ifdef __has_builtin
# if __has_builtin(__builtin_amdgcn_tensor_load_to_lds) && __has_builtin(__builtin_amdgcn_s_wait_tensorcnt)
#  define USE_TDM 1
# endif
#endif
#ifndef USE_TDM
# define USE_TDM 0
#endif

union AFrag { uint4 u[2]; v16h v; };

static __device__ __forceinline__ v16h load_v16h(const _Float16* p) {
    AFrag t;
    t.u[0] = *(const uint4*)p;
    t.u[1] = *(const uint4*)(p + 8);
    return t.v;
}

static __device__ __forceinline__ v8f wmma_f16(v16h a, v16h b, v8f c) {
    // D = A(16x32) * B(32x16) + C, f32 accumulate
    return __builtin_amdgcn_wmma_f32_16x16x32_f16(
        /*neg_a=*/false, a, /*neg_b=*/false, b,
        /*c_mod=*/(short)0, c, /*reuse_a=*/false, /*reuse_b=*/false);
}

#if USE_TDM
// ---------------------------------------------------------------------------
// TDM: DMA a 128-row x 32-f16 tile (row stride `kstride` elements in memory)
// into LDS at `lds`, inserting 8 DWORDs of pad after every 16 DWORDs so each
// 64B row lands on a 96B (48-element) stride -> bank-skewed LDS layout.
// D# layout per CDNA5 ISA section 8.3/8.4.
// ---------------------------------------------------------------------------
static __device__ __forceinline__ void tdm_load_tile_128x32(void* lds,
                                                            const _Float16* gsrc,
                                                            int kstride) {
    const unsigned long long ga = (unsigned long long)(uintptr_t)gsrc;
    const unsigned int       la = (unsigned int)(uintptr_t)lds;
    u32x4 g0;
    g0[0] = 1u;                                    // count=1, user mode, no gather
    g0[1] = la;                                    // lds_addr [63:32]
    g0[2] = (unsigned int)ga;                      // global_addr [95:64]
    g0[3] = (unsigned int)(ga >> 32) | (2u << 30); // global_addr hi + type=2
    i32x8 g1;
    // data_size=1(2B) | pad_enable | pad_interval=3(16 DW) | pad_amount=7(8 DW)
    g1[0] = (int)((1u << 16) | (1u << 20) | (3u << 22) | (7u << 25));
    g1[1] = (int)((kstride & 0xFFFF) << 16);       // tensor_dim0 lo16 @ bits[63:48]
    g1[2] = (int)((kstride >> 16) & 0xFFFF);       // tensor_dim0 hi16 (dim1 lo16 = 0)
    g1[3] = (int)(16 | (32u << 16));               // tensor_dim1 hi16 (=1M) | tile_dim0=32
    g1[4] = 128;                                   // tile_dim1=128, tile_dim2=0
    g1[5] = kstride;                               // tensor_dim0_stride lo32
    g1[6] = 0;                                     // stride hi16 | dim1_stride lo16
    g1[7] = 0;                                     // dim1_stride hi32
    const i32x4 z4 = {0, 0, 0, 0};
#if defined(__clang_major__) && (__clang_major__ >= 23)
    const i32x8 z8 = {0, 0, 0, 0, 0, 0, 0, 0};
    __builtin_amdgcn_tensor_load_to_lds(g0, g1, z4, z4, z8, 0);
#else
    __builtin_amdgcn_tensor_load_to_lds(g0, g1, z4, z4, 0);
#endif
}
#endif // USE_TDM

// ---------------------------------------------------------------------------
// fp32 -> f16 elementwise convert
// ---------------------------------------------------------------------------
__global__ void cvt_f32_f16(const float* __restrict__ in, _Float16* __restrict__ out) {
    size_t i = ((size_t)blockIdx.x * blockDim.x + threadIdx.x) * 4;
    float4 v = *(const float4*)(in + i);
    union { _Float16 h[4]; uint2 u; } p;
    p.h[0] = (_Float16)v.x; p.h[1] = (_Float16)v.y;
    p.h[2] = (_Float16)v.z; p.h[3] = (_Float16)v.w;
    *(uint2*)(out + i) = p.u;
}

// ---------------------------------------------------------------------------
// fp32 [R][C] -> f16 [C][R] transposed convert, LDS-tiled (per batch in .z)
// ---------------------------------------------------------------------------
__global__ void tcvt_f32_f16(const float* __restrict__ in, _Float16* __restrict__ out,
                             int R, int C) {
    __shared__ float tile[32][33];
    const size_t mo = (size_t)blockIdx.z * (size_t)R * (size_t)C;
    in  += mo;
    out += mo;
    const int cb = blockIdx.x * 32, rb = blockIdx.y * 32;
    const int tx = threadIdx.x, ty = threadIdx.y;
#pragma unroll
    for (int j = 0; j < 4; j++)
        tile[ty + 8*j][tx] = in[(size_t)(rb + ty + 8*j) * C + cb + tx];
    __syncthreads();
#pragma unroll
    for (int j = 0; j < 4; j++)
        out[(size_t)(cb + ty + 8*j) * R + rb + tx] = (_Float16)tile[tx][ty + 8*j];
}

// ---------------------------------------------------------------------------
// C[M,N] = A[M,K] (f16, row major) * B (given as BT[N,K], f16, row major)
// block: 256 thr = 8 waves in a 4x2 grid; block tile 128x128, BK=32.
// wave tile = 32 rows x 64 cols: 2 A-frags + 4 B-frags feed 8 WMMAs per slab.
// A/BT tiles staged into LDS by the Tensor Data Mover (pad feature builds the
// 48-element skewed row stride); sync-staging fallback if builtin is absent.
// ---------------------------------------------------------------------------
template <bool OUT_F32>
__global__ __launch_bounds__(256) void gemm_f16(const _Float16* __restrict__ A,
                                                const _Float16* __restrict__ BT,
                                                void* __restrict__ Cp,
                                                int M, int N, int K) {
    constexpr int LS = 48;                 // padded LDS row stride (96B, 16B aligned)
    __shared__ _Float16 As[128 * LS];
    __shared__ _Float16 Bs[128 * LS];

    const int t  = threadIdx.x;
    const int w  = t >> 5, l = t & 31;
    const int ln = l & 15, kh = l >> 4;    // fragment lane coords per ISA layout
    const int wr = w >> 1;                 // wave row group: 0..3 (32 rows each)
    const int wc = w & 1;                  // wave col group: 0..1 (64 cols each)
    const int mbase = blockIdx.x * 128, nbase = blockIdx.y * 128;

    v8f acc[2][4];
#pragma unroll
    for (int i = 0; i < 2; i++)
#pragma unroll
        for (int j = 0; j < 4; j++) acc[i][j] = {};

    const int nkb = K >> 5;
    for (int kb = 0; kb < nkb; kb++) {
#if USE_TDM
        if (t == 0) {                      // one wave drives the DMA engine
            tdm_load_tile_128x32(As, &A [(size_t)mbase * K + kb * 32], K);
            tdm_load_tile_128x32(Bs, &BT[(size_t)nbase * K + kb * 32], K);
            __builtin_amdgcn_s_wait_tensorcnt(0);
        }
#else
        // ---- cooperative staging: 512 chunks of 8 f16, coalesced b128 ----
#pragma unroll
        for (int q = 0; q < 2; q++) {
            int cid = t + q * 256;
            int row = cid >> 2, ko = (cid & 3) << 3;
            *(uint4*)&As[row * LS + ko] =
                *(const uint4*)&A[(size_t)(mbase + row) * K + kb * 32 + ko];
            *(uint4*)&Bs[row * LS + ko] =
                *(const uint4*)&BT[(size_t)(nbase + row) * K + kb * 32 + ko];
        }
#endif
        if (kb + 1 < nkb) {                // gfx1250 global_prefetch_b8 (L2 warm)
            int row = t >> 1, ko = (t & 1) << 4;
            __builtin_prefetch(&A [(size_t)(mbase + row) * K + (kb + 1) * 32 + ko], 0, 3);
            __builtin_prefetch(&BT[(size_t)(nbase + row) * K + (kb + 1) * 32 + ko], 0, 3);
        }
        __syncthreads();

        // ---- two A fragments: wave's 32 rows of this 32-K slab ----
        AFrag a0, a1;
        {
            const _Float16* ap0 = &As[(wr * 32 + ln) * LS];
            const _Float16* ap1 = &As[(wr * 32 + 16 + ln) * LS];
            a0.u[0] = *(const uint4*)(ap0 + kh * 16);
            a0.u[1] = *(const uint4*)(ap0 + 8 + kh * 16);
            a1.u[0] = *(const uint4*)(ap1 + kh * 16);
            a1.u[1] = *(const uint4*)(ap1 + 8 + kh * 16);
        }
#pragma unroll
        for (int ct = 0; ct < 4; ct++) {
            v16h b = load_v16h(&Bs[(wc * 64 + ct * 16 + ln) * LS + kh * 16]);
            acc[0][ct] = wmma_f16(a0.v, b, acc[0][ct]);
            acc[1][ct] = wmma_f16(a1.v, b, acc[1][ct]);
        }
        __syncthreads();
    }

    // ---- epilogue: C layout lane(m = v + 8*kh, n = ln) ----
    const int m00 = mbase + wr * 32 + 8 * kh;
#pragma unroll
    for (int rt = 0; rt < 2; rt++) {
#pragma unroll
        for (int ct = 0; ct < 4; ct++) {
            const int col = nbase + wc * 64 + ct * 16 + ln;
#pragma unroll
            for (int v = 0; v < 8; v++) {
                const size_t idx = (size_t)(m00 + rt * 16 + v) * N + col;
                if constexpr (OUT_F32) ((float*)Cp)[idx] = acc[rt][ct][v];
                else                   ((_Float16*)Cp)[idx] = (_Float16)acc[rt][ct][v];
            }
        }
    }
    (void)M;
}

// ---------------------------------------------------------------------------
// Flash attention: one block per (16-query tile, batch). 256 thr = 8 waves.
//  qf  : [4*2048][1024] f16  (q = x @ wqk)
//  xf  : [4*2048][1024] f16  (keys/values, row major)
//  xfT : [4][1024][2048] f16 (values transposed: d-major)
//  ctx : [4*2048][1024] f16 output
// ---------------------------------------------------------------------------
__global__ __launch_bounds__(256) void attn_flash(const _Float16* __restrict__ qf,
                                                  const _Float16* __restrict__ xf,
                                                  const _Float16* __restrict__ xfT,
                                                  _Float16* __restrict__ ctx) {
    constexpr int QS = 1032;   // q tile LDS stride (pad: 16B aligned, bank-skewed)
    constexpr int SS = 264;    // score/P LDS stride
    __shared__ _Float16 qs[16 * QS];
    __shared__ float    Sb[16 * SS];
    __shared__ _Float16 Pb[16 * SS];
    __shared__ float    m_run[16], l_run[16], sc[16];

    const int t  = threadIdx.x;
    const int w  = t >> 5, l = t & 31;
    const int ln = l & 15, kh = l >> 4;
    const int bi = blockIdx.y, ti = blockIdx.x;
    const int q0 = ti * 16;
    const size_t xoff  = (size_t)bi * 2048 * 1024;   // xf / qf / ctx batch offset
    const size_t xtoff = (size_t)bi * 1024 * 2048;   // xfT batch offset

    if (t < 16) { m_run[t] = NEG_INF; l_run[t] = 0.0f; }
    // load q tile (16 x 1024) into padded LDS
#pragma unroll
    for (int i = 0; i < 8; i++) {
        int cid = t + i * 256;                       // 2048 chunks of 8 f16
        int row = cid >> 7, ko = (cid & 127) << 3;
        *(uint4*)&qs[row * QS + ko] =
            *(const uint4*)&qf[xoff + (size_t)(q0 + row) * 1024 + ko];
    }
    __syncthreads();

    v8f acc[8];
#pragma unroll
    for (int i = 0; i < 8; i++) acc[i] = {};

    const int r  = t >> 4, sl = t & 15;              // softmax: 16 lanes per row
    const int nkb = (q0 + 16 + 255) >> 8;            // 256-key blocks over causal span

    for (int kb0 = 0; kb0 < nkb; kb0++) {
        const int keybase = kb0 << 8;

        // ---- S block: each wave computes 2 of 16 column tiles ----
        v8f s0 = {}, s1 = {};
        const int cb0 = keybase + (2 * w) * 16;
        const int cb1 = cb0 + 16;
        const _Float16* kp0 = xf + xoff + (size_t)(cb0 + ln) * 1024 + kh * 16;
        const _Float16* kp1 = xf + xoff + (size_t)(cb1 + ln) * 1024 + kh * 16;
        if (kb0 + 1 < nkb) {
            __builtin_prefetch(kp0 + (size_t)256 * 1024, 0, 3);
            __builtin_prefetch(kp1 + (size_t)256 * 1024, 0, 3);
        }
#pragma unroll 8
        for (int kk = 0; kk < 32; kk++) {
            AFrag af;
            const _Float16* qp = &qs[ln * QS + kk * 32];
            af.u[0] = *(const uint4*)(qp + kh * 16);
            af.u[1] = *(const uint4*)(qp + 8 + kh * 16);
            v16h b0 = load_v16h(kp0 + kk * 32);
            v16h b1 = load_v16h(kp1 + kk * 32);
            s0 = wmma_f16(af.v, b0, s0);
            s1 = wmma_f16(af.v, b1, s1);
        }
        // causal mask + store (C layout: m = v + 8*kh, n = ln)
#pragma unroll
        for (int v = 0; v < 8; v++) {
            const int m = v + 8 * kh, row = q0 + m;
            Sb[m * SS + (2 * w) * 16 + ln]     = (cb0 + ln <= row) ? s0[v] : NEG_INF;
            Sb[m * SS + (2 * w + 1) * 16 + ln] = (cb1 + ln <= row) ? s1[v] : NEG_INF;
        }
        __syncthreads();

        // ---- online softmax over this 256-key block ----
        {
            float bm = NEG_INF;
#pragma unroll 4
            for (int c = sl; c < 256; c += 16) bm = fmaxf(bm, Sb[r * SS + c]);
#pragma unroll
            for (int off = 8; off > 0; off >>= 1) bm = fmaxf(bm, __shfl_xor(bm, off, 16));
            const float mo = m_run[r];
            const float mn = fmaxf(mo, bm);
            float bsum = 0.0f;
#pragma unroll 4
            for (int c = sl; c < 256; c += 16) {
                float e = __expf(Sb[r * SS + c] - mn);
                Pb[r * SS + c] = (_Float16)e;     // unnormalized P
                bsum += e;
            }
#pragma unroll
            for (int off = 8; off > 0; off >>= 1) bsum += __shfl_xor(bsum, off, 16);
            if (sl == 0) {
                const float scale = __expf(mo - mn);
                sc[r]    = scale;
                m_run[r] = mn;
                l_run[r] = l_run[r] * scale + bsum;
            }
        }
        __syncthreads();

        // ---- rescale acc, then acc += P(16x256) @ V(256 x 128-slice) ----
        v8f scv;
#pragma unroll
        for (int v = 0; v < 8; v++) scv[v] = sc[v + 8 * kh];
#pragma unroll
        for (int ct = 0; ct < 8; ct++) acc[ct] *= scv;

        const int d0 = w * 128;
#pragma unroll
        for (int ks = 0; ks < 8; ks++) {
            AFrag af;
            const _Float16* pp = &Pb[ln * SS + ks * 32];
            af.u[0] = *(const uint4*)(pp + kh * 16);
            af.u[1] = *(const uint4*)(pp + 8 + kh * 16);
#pragma unroll
            for (int ct = 0; ct < 8; ct++) {
                const _Float16* vp = xfT + xtoff +
                    (size_t)(d0 + ct * 16 + ln) * 2048 + keybase + ks * 32 + kh * 16;
                v16h b = load_v16h(vp);
                acc[ct] = wmma_f16(af.v, b, acc[ct]);
            }
        }
        __syncthreads();
    }

    // ---- normalize by running sum and write ctx (f16) ----
    v8f inv;
#pragma unroll
    for (int v = 0; v < 8; v++) inv[v] = 1.0f / l_run[v + 8 * kh];
#pragma unroll
    for (int ct = 0; ct < 8; ct++) {
        const int col = w * 128 + ct * 16 + ln;
#pragma unroll
        for (int v = 0; v < 8; v++) {
            const int row = q0 + v + 8 * kh;
            ctx[xoff + (size_t)row * 1024 + col] = (_Float16)(acc[ct][v] * inv[v]);
        }
    }
}

// ---------------------------------------------------------------------------
extern "C" void kernel_launch(void* const* d_in, const int* in_sizes, int n_in,
                              void* d_out, int out_size, void* d_ws, size_t ws_size,
                              hipStream_t stream) {
    (void)in_sizes; (void)n_in; (void)out_size; (void)ws_size;
    const float* x   = (const float*)d_in[0];   // [4][2048][1024]
    const float* wqk = (const float*)d_in[1];   // [1024][1024]
    const float* wov = (const float*)d_in[2];   // [1024][1024]
    float* out = (float*)d_out;                 // [4][2048][1024]

    const int NTOK = 4 * 2048;                  // 8192 rows flattened
    const int D    = 1024;

    char* ws = (char*)d_ws;
    size_t off = 0;
    _Float16* xb   = (_Float16*)(ws + off); off += (size_t)NTOK * D * 2;  // x f16
    _Float16* xbT  = (_Float16*)(ws + off); off += (size_t)NTOK * D * 2;  // x^T per batch
    _Float16* qbuf = (_Float16*)(ws + off); off += (size_t)NTOK * D * 2;  // q = x@wqk
    _Float16* ctxb = (_Float16*)(ws + off); off += (size_t)NTOK * D * 2;  // attn@x
    _Float16* wqkT = (_Float16*)(ws + off); off += (size_t)D * D * 2;
    _Float16* wovT = (_Float16*)(ws + off); off += (size_t)D * D * 2;

    // fp32 -> f16 (plus transposed copies for K-contiguous WMMA B fragments)
    cvt_f32_f16<<<(NTOK * D / 4) / 256, 256, 0, stream>>>(x, xb);
    tcvt_f32_f16<<<dim3(D / 32, 2048 / 32, 4), dim3(32, 8), 0, stream>>>(x, xbT, 2048, D);
    tcvt_f32_f16<<<dim3(D / 32, D / 32, 1), dim3(32, 8), 0, stream>>>(wqk, wqkT, D, D);
    tcvt_f32_f16<<<dim3(D / 32, D / 32, 1), dim3(32, 8), 0, stream>>>(wov, wovT, D, D);

    // q = x @ wqk  (f16 out)
    gemm_f16<false><<<dim3(NTOK / 128, D / 128), 256, 0, stream>>>(xb, wqkT, qbuf, NTOK, D, D);

    // ctx = softmax(causal(q @ x^T)) @ x   (flash, f16 out)
    attn_flash<<<dim3(2048 / 16, 4), 256, 0, stream>>>(qbuf, xb, xbT, ctxb);

    // out = ctx @ wov  (f32 out)
    gemm_f16<true><<<dim3(NTOK / 128, D / 128), 256, 0, stream>>>(ctxb, wovT, out, NTOK, D, D);
}
